// KV_Cross_Attention_E2P_43344809952061
// MI455X (gfx1250) — matrix operand order repaired and checked by reading, not compiled, not run on hardware
//
#include <hip/hip_runtime.h>
#include <hip/hip_bf16.h>
#include <stdint.h>

// ---------------- problem constants ----------------
#define T_  4
#define B_  16
#define C_  384      // = M = K of both GEMMs
#define N_  1024     // H*W
#define STRIDE_ (B_ * C_ * N_)   // elements per time step

typedef __attribute__((ext_vector_type(16))) _Float16 v16h;
typedef __attribute__((ext_vector_type(8)))  _Float16 v8h;
typedef __attribute__((ext_vector_type(8)))  float    v8f;

union HV16 { v16h v; v8h h[2]; };

// ---------------- kernel 0a: fold BN (+bias) into per-channel affine --------------
__global__ void bn_prep(const float* __restrict__ qg, const float* __restrict__ qb,
                        const float* __restrict__ qm, const float* __restrict__ qv,
                        const float* __restrict__ pg, const float* __restrict__ pb,
                        const float* __restrict__ pm, const float* __restrict__ pv,
                        const float* __restrict__ bp,
                        float* __restrict__ qscale, float* __restrict__ qshift,
                        float* __restrict__ pscale, float* __restrict__ pshift) {
  int i = blockIdx.x * blockDim.x + threadIdx.x;
  if (i >= C_) return;
  float invq = qg[i] * __frsqrt_rn(qv[i] + 1e-5f);
  qscale[i] = invq;
  qshift[i] = qb[i] - qm[i] * invq;
  float invp = pg[i] * __frsqrt_rn(pv[i] + 1e-5f);
  pscale[i] = invp;
  // out = (y + bp)*invp + (beta - mean*invp)
  pshift[i] = bp[i] * invp + pb[i] - pm[i] * invp;
}

// ---------------- kernel 0b: pre-convert both weight matrices fp32 -> fp16 -------
__global__ void w_to_h(const float* __restrict__ wq, const float* __restrict__ wp,
                       _Float16* __restrict__ wqh, _Float16* __restrict__ wph) {
  int i = blockIdx.x * blockDim.x + threadIdx.x;
  if (i >= C_ * C_) return;
  wqh[i] = (_Float16)wq[i];
  wph[i] = (_Float16)wp[i];
}

// ---------------- kernel 1: LIF over x -> fp16 spikes, transposed to (T,B,N,C) ----
// tau=2, v_th=1, hard reset to 0. Block = 32c x 32n tile of one batch b; LIF scan
// over T in registers; padded-LDS transpose so both read and write are coalesced.
__global__ __launch_bounds__(256) void lif_x_t(const float* __restrict__ x,
                                               _Float16* __restrict__ xs) {
  __shared__ _Float16 tile[T_][32][33];
  const int b  = blockIdx.z;
  const int c0 = blockIdx.y * 32;
  const int n0 = blockIdx.x * 32;
  const int tx = threadIdx.x & 31;
  const int ty = threadIdx.x >> 5;   // 0..7
  float v[4] = {0.f, 0.f, 0.f, 0.f};
#pragma unroll
  for (int t = 0; t < T_; ++t) {
#pragma unroll
    for (int r = 0; r < 4; ++r) {
      int cl = ty + 8 * r;
      float xv = x[(((size_t)t * B_ + b) * C_ + c0 + cl) * N_ + n0 + tx];
      float h  = v[r] + (xv - v[r]) * 0.5f;
      float s  = (h - 1.0f) >= 0.f ? 1.f : 0.f;
      v[r] = h * (1.f - s);
      tile[t][cl][tx] = (_Float16)s;
    }
  }
  __syncthreads();
#pragma unroll
  for (int t = 0; t < T_; ++t) {
#pragma unroll
    for (int r = 0; r < 4; ++r) {
      int nl = ty + 8 * r;
      xs[(((size_t)t * B_ + b) * N_ + n0 + nl) * C_ + c0 + tx] = tile[t][tx][nl];
    }
  }
}

// ---------------- kernel 3: LIF over q + (attn_k & attn_v) mask, transposed -----
// q is (T,B,C,N) fp32 from GEMM1; masks are (T,B,C,N) bytes; output (T,B,N,C) fp16.
__global__ __launch_bounds__(256) void lif_q_mask_t(const float* __restrict__ q,
                                                    const uint8_t* __restrict__ ak,
                                                    const uint8_t* __restrict__ av,
                                                    _Float16* __restrict__ s) {
  __shared__ _Float16 tile[T_][32][33];
  const int b  = blockIdx.z;
  const int c0 = blockIdx.y * 32;
  const int n0 = blockIdx.x * 32;
  const int tx = threadIdx.x & 31;
  const int ty = threadIdx.x >> 5;
  float v[4] = {0.f, 0.f, 0.f, 0.f};
#pragma unroll
  for (int t = 0; t < T_; ++t) {
#pragma unroll
    for (int r = 0; r < 4; ++r) {
      int cl = ty + 8 * r;
      size_t off = (((size_t)t * B_ + b) * C_ + c0 + cl) * N_ + n0 + tx;
      float qv = q[off];
      float h  = v[r] + (qv - v[r]) * 0.5f;
      float sp = (h - 1.0f) >= 0.f ? 1.f : 0.f;
      v[r] = h * (1.f - sp);
      float m = (ak[off] != 0 && av[off] != 0) ? sp : 0.f;
      tile[t][cl][tx] = (_Float16)m;
    }
  }
  __syncthreads();
#pragma unroll
  for (int t = 0; t < T_; ++t) {
#pragma unroll
    for (int r = 0; r < 4; ++r) {
      int nl = ty + 8 * r;
      s[(((size_t)t * B_ + b) * N_ + n0 + nl) * C_ + c0 + tx] = tile[t][tx][nl];
    }
  }
}

// ---------------- GEMM + fused channel affine ----------------
// D[tb][m][n] = (sum_k W[m][k] * S[tb][n][k]) * scale[m] + shift[m]
// W pre-converted fp16 [m][k]; S activation-transposed (T*B, N, C) fp16 [n][k].
// Both tiles are pure b128 global->LDS copies (no scatter, no cvt in the loop).
// Block: 256 threads = 8 waves (wave32), 128x128 output tile.
// Wave grid 2(M) x 4(N); each wave owns 64x32 = 4x2 WMMA 16x16 f32 accumulators.
#define BM 128
#define BN 128
#define KC 32
#define LDK 40   // padded LDS row stride (halves)

__global__ __launch_bounds__(256) void gemm_bn(
    const _Float16* __restrict__ Wh,     // [C_][C_] fp16 row-major
    const _Float16* __restrict__ Sg,     // [T*B][N_][C_] fp16 (n-major rows)
    const float*    __restrict__ scale,  // [C_]
    const float*    __restrict__ shift,  // [C_]
    float*          __restrict__ Og) {   // [T*B][C_][N_] fp32
  __shared__ __align__(16) _Float16 As[BM * LDK];
  __shared__ __align__(16) _Float16 Bs[BN * LDK];

  const int tb = blockIdx.z;
  const _Float16* Bsrc = Sg + (size_t)tb * N_ * C_;
  float*          Out  = Og + (size_t)tb * C_ * N_;

  const int m0 = blockIdx.y * BM;
  const int n0 = blockIdx.x * BN;
  const int tid  = threadIdx.x;
  const int lane = tid & 31;
  const int wave = tid >> 5;
  const int wm = wave >> 2;   // 0..1
  const int wn = wave & 3;    // 0..3
  const int hi = lane >> 4;   // lane half-group (0 or 1)
  const int lr = lane & 15;

  v8f acc[4][2];
#pragma unroll
  for (int i = 0; i < 4; ++i)
#pragma unroll
    for (int j = 0; j < 2; ++j) acc[i][j] = (v8f){0.f,0.f,0.f,0.f,0.f,0.f,0.f,0.f};

  const int row = tid >> 1;          // 0..127 (shared by A and B loaders)
  const int seg = (tid & 1) * 16;    // 0 or 16 halves along k

  for (int kc = 0; kc < C_; kc += KC) {
    __syncthreads();
    {
      // A tile: 128 m x 32 k fp16, k-contiguous in global.
      const _Float16* src = Wh + (size_t)(m0 + row) * C_ + kc + seg;
      _Float16* dst = &As[row * LDK + seg];
      *(v8h*)(dst)     = *(const v8h*)(src);
      *(v8h*)(dst + 8) = *(const v8h*)(src + 8);
    }
    {
      // B tile: 128 n x 32 k fp16, k-contiguous in global.
      const _Float16* src = Bsrc + (size_t)(n0 + row) * C_ + kc + seg;
      _Float16* dst = &Bs[row * LDK + seg];
      *(v8h*)(dst)     = *(const v8h*)(src);
      *(v8h*)(dst + 8) = *(const v8h*)(src + 8);
    }
    __syncthreads();

    // B fragments: lane = n column; 16 contiguous K per lane-half.
    HV16 bf[2];
#pragma unroll
    for (int j = 0; j < 2; ++j) {
      const _Float16* p = &Bs[(wn * 32 + j * 16 + lr) * LDK + hi * 16];
      bf[j].h[0] = *(const v8h*)(p);
      bf[j].h[1] = *(const v8h*)(p + 8);
    }
#pragma unroll
    for (int i = 0; i < 4; ++i) {
      // A fragment: lane = m row; 16 contiguous K per lane-half (K=16*hi+e).
      const _Float16* p = &As[(wm * 64 + i * 16 + lr) * LDK];
      HV16 af;
      af.h[0] = *(const v8h*)(p + 16 * hi);
      af.h[1] = *(const v8h*)(p + 16 * hi + 8);
#pragma unroll
      for (int j = 0; j < 2; ++j) {
        acc[i][j] = __builtin_amdgcn_wmma_f32_16x16x32_f16(
            false, af.v, false, bf[j].v, (short)0, acc[i][j], false, false);
      }
    }
  }

  // Epilogue: C/D layout — lane<16: m=r, n=lane ; lane>=16: m=r+8, n=lane-16.
#pragma unroll
  for (int i = 0; i < 4; ++i) {
    const int mbase = m0 + wm * 64 + i * 16 + 8 * hi;
#pragma unroll
    for (int r = 0; r < 8; ++r) {
      const int m = mbase + r;
      const float sc = scale[m], sh = shift[m];
#pragma unroll
      for (int j = 0; j < 2; ++j) {
        const int n = n0 + wn * 32 + j * 16 + lr;
        Out[(size_t)m * N_ + n] = acc[i][j][r] * sc + sh;
      }
    }
  }
}

// ---------------- launcher ----------------
extern "C" void kernel_launch(void* const* d_in, const int* in_sizes, int n_in,
                              void* d_out, int out_size, void* d_ws, size_t ws_size,
                              hipStream_t stream) {
  const float*   x   = (const float*)d_in[0];
  const uint8_t* ak  = (const uint8_t*)d_in[1];
  const uint8_t* av  = (const uint8_t*)d_in[2];
  const float*   wq  = (const float*)d_in[3];
  const float*   qg  = (const float*)d_in[4];
  const float*   qb  = (const float*)d_in[5];
  const float*   qm  = (const float*)d_in[6];
  const float*   qv  = (const float*)d_in[7];
  const float*   wp  = (const float*)d_in[8];
  const float*   bp  = (const float*)d_in[9];
  const float*   pg  = (const float*)d_in[10];
  const float*   pb  = (const float*)d_in[11];
  const float*   pm  = (const float*)d_in[12];
  const float*   pv  = (const float*)d_in[13];
  float* out = (float*)d_out;

  // workspace layout
  char* ws = (char*)d_ws;
  size_t off = 0;
  _Float16* xs = (_Float16*)(ws + off); off += (size_t)T_ * STRIDE_ * sizeof(_Float16); // (T,B,N,C)
  off = (off + 255) & ~(size_t)255;
  float* qbuf  = (float*)(ws + off);    off += (size_t)T_ * STRIDE_ * sizeof(float);    // (T,B,C,N)
  off = (off + 255) & ~(size_t)255;
  _Float16* sm = (_Float16*)(ws + off); off += (size_t)T_ * STRIDE_ * sizeof(_Float16); // (T,B,N,C)
  off = (off + 255) & ~(size_t)255;
  _Float16* wqh = (_Float16*)(ws + off); off += (size_t)C_ * C_ * sizeof(_Float16);
  off = (off + 255) & ~(size_t)255;
  _Float16* wph = (_Float16*)(ws + off); off += (size_t)C_ * C_ * sizeof(_Float16);
  off = (off + 255) & ~(size_t)255;
  float* qscale = (float*)(ws + off); off += C_ * sizeof(float);
  float* qshift = (float*)(ws + off); off += C_ * sizeof(float);
  float* pscale = (float*)(ws + off); off += C_ * sizeof(float);
  float* pshift = (float*)(ws + off); off += C_ * sizeof(float);
  (void)ws_size; (void)in_sizes; (void)n_in; (void)out_size;

  // 0: BN/bias folding + weight fp16 conversion
  bn_prep<<<(C_ + 255) / 256, 256, 0, stream>>>(qg, qb, qm, qv, pg, pb, pm, pv, bp,
                                                qscale, qshift, pscale, pshift);
  w_to_h<<<(C_ * C_ + 255) / 256, 256, 0, stream>>>(wq, wp, wqh, wph);

  dim3 egrid(N_ / 32, C_ / 32, B_);        // (32, 12, 16) tiles for elementwise+transpose
  // 1: shortcut LIF -> fp16 spikes (T,B,N,C)
  lif_x_t<<<egrid, 256, 0, stream>>>(x, xs);

  // 2: q = wq . xs, fused q_bn -> (T,B,C,N) fp32
  dim3 ggrid(N_ / BN, C_ / BM, T_ * B_);   // (8, 3, 64)
  gemm_bn<<<ggrid, 256, 0, stream>>>(wqh, xs, qscale, qshift, qbuf);

  // 3: q_lif + (attn_k & attn_v) mask -> fp16 (T,B,N,C)
  lif_q_mask_t<<<egrid, 256, 0, stream>>>(qbuf, ak, av, sm);

  // 4: out = wp . masked + bias, fused proj_bn -> d_out (T,B,C,H,W)
  gemm_bn<<<ggrid, 256, 0, stream>>>(wph, sm, pscale, pshift, out);
}
